// SpatialGCN_274877907544
// MI455X (gfx1250) — matrix operand order repaired and checked by reading, not compiled, not run on hardware
//
#include <hip/hip_runtime.h>
#include <hip/hip_bf16.h>

// ---------------------------------------------------------------------------
// SpatialGCN forward, fused for gfx1250 (MI455X), f16 WMMA + f32 accumulate.
//
// Math (per batch b, N = h*w = 16384, C=256 plane, I=128 inter):
//   k,q,v = W{k,q,v} @ x + b        (fused as one 384x256 GEMM, K=256)
//   S     = q @ v^T                 (K=16384, split into 32 K-slices -> Spart)
//   A     = softmax_d(S)
//   T     = w_wg @ A^T              (stored transposed: Tt[c][o])
//   M2    = P1 @ T,  P1 = diag(s2) w_out diag(s_wg)      [256 x 128]
//   out   = relu(M2 @ k + c1 + x),  c1 = s2*(w_out@shift_wg + b_out)+shift2
//
// All GEMMs are in NT form (both operands row-major, K contiguous) so one
// fragment-load pattern serves every kernel:
//   A-frag (16x32 f16): lane l -> row m0+(l&15), K = k0+(l>>4)*8 .. (+16 ..)
//   B-frag (32x16 f16): lane l -> col n0+(l&15), K = k0+(l>>4)*8 .. (+16 ..)
//   C/D    (16x16 f32): lane l -> col l&15, rows r+(l>>4)*8
//
// f16-tile staging into LDS uses GLOBAL_LOAD_ASYNC_TO_LDS_B128 (ASYNCcnt)
// when the toolchain exposes the builtin; otherwise a sync b128 copy.
// ---------------------------------------------------------------------------

#define DEVI static __device__ __forceinline__

typedef __attribute__((ext_vector_type(16))) _Float16 v16h;
typedef __attribute__((ext_vector_type(8)))  _Float16 v8h;
typedef __attribute__((ext_vector_type(8)))  float    v8f;
typedef int i4 __attribute__((vector_size(16)));

union F16x16 { v16h h; uint4 u[2]; };
union F16x8  { v8h  h; uint4 u;    };

#define BATCH 8
#define PLANE 256
#define INTER 128
#define NPIX  16384

#if defined(__gfx1250__) && __has_builtin(__builtin_amdgcn_global_load_async_to_lds_b128)
#define USE_ASYNC_LDS 1
#else
#define USE_ASYNC_LDS 0
#endif

DEVI v8f vzero8() {
  v8f v;
#pragma unroll
  for (int i = 0; i < 8; ++i) v[i] = 0.0f;
  return v;
}

// 16-byte global -> LDS copy (async data mover when available).
DEVI void stage16(const void* g, void* l) {
#if USE_ASYNC_LDS
  typedef __attribute__((address_space(1))) i4 gi4;
  typedef __attribute__((address_space(3))) i4 li4;
  __builtin_amdgcn_global_load_async_to_lds_b128((gi4*)const_cast<void*>(g),
                                                 (li4*)l, 0, 0);
#else
  *(uint4*)l = *(const uint4*)g;
#endif
}
DEVI void stage_wait() {
#if USE_ASYNC_LDS
#if __has_builtin(__builtin_amdgcn_s_wait_asynccnt)
  __builtin_amdgcn_s_wait_asynccnt(0);
#else
  asm volatile("s_wait_asynccnt 0x0" ::: "memory");
#endif
#endif
}

// NT A-fragment: base row-major [*, ld], rows m0..m0+15, K-block k0..k0+31
DEVI v16h frag_a(const _Float16* base, int ld, int m0, int k0, int lane) {
  const _Float16* p = base + (size_t)(m0 + (lane & 15)) * ld + k0 + ((lane >> 4) << 3);
  F16x16 f;
  f.u[0] = *(const uint4*)(p);
  f.u[1] = *(const uint4*)(p + 16);
  return f.h;
}
// NT B-fragment: base row-major over n [*, ld], cols n0..n0+15, K-block k0..k0+31
DEVI v16h frag_b(const _Float16* base, int ld, int n0, int k0, int lane) {
  const _Float16* p = base + (size_t)(n0 + (lane & 15)) * ld + k0 + ((lane >> 4) << 3);
  F16x16 f;
  f.u[0] = *(const uint4*)(p);
  f.u[1] = *(const uint4*)(p + 16);
  return f.h;
}
DEVI v8f wmma_f16(v16h a, v16h b, v8f c) {
  return __builtin_amdgcn_wmma_f32_16x16x32_f16(false, a, false, b, (short)0, c,
                                                false, false);
}

// ---------------------------------------------------------------------------
// prep1: pack weights to f16, fold BN constants
// ---------------------------------------------------------------------------
__global__ void k_prep1(const float* __restrict__ wk, const float* __restrict__ wq,
                        const float* __restrict__ wv, const float* __restrict__ bk,
                        const float* __restrict__ bq, const float* __restrict__ bv,
                        const float* __restrict__ wwg_f, const float* __restrict__ g_wg,
                        const float* __restrict__ beta_wg, const float* __restrict__ m_wg,
                        const float* __restrict__ v_wg, const float* __restrict__ g_out,
                        const float* __restrict__ beta_out, const float* __restrict__ m_out,
                        const float* __restrict__ v_out,
                        _Float16* __restrict__ Wall, _Float16* __restrict__ wwg16,
                        float* __restrict__ ball, float* __restrict__ scale_wg,
                        float* __restrict__ shift_wg, float* __restrict__ scale2,
                        float* __restrict__ shift2) {
  const int gid = blockIdx.x * blockDim.x + threadIdx.x;
  const int stride = gridDim.x * blockDim.x;
  for (int i = gid; i < 384 * 256; i += stride) {
    float w = (i < 32768) ? wk[i] : (i < 65536) ? wq[i - 32768] : wv[i - 65536];
    Wall[i] = (_Float16)w;
  }
  for (int i = gid; i < 128 * 128; i += stride) wwg16[i] = (_Float16)wwg_f[i];
  for (int i = gid; i < 384; i += stride)
    ball[i] = (i < 128) ? bk[i] : (i < 256) ? bq[i - 128] : bv[i - 256];
  for (int i = gid; i < 128; i += stride) {
    float sc = g_wg[i] * rsqrtf(v_wg[i] + 1e-5f);
    scale_wg[i] = sc;
    shift_wg[i] = beta_wg[i] - m_wg[i] * sc;
  }
  for (int i = gid; i < 256; i += stride) {
    float sc = g_out[i] * rsqrtf(v_out[i] + 1e-5f);
    scale2[i] = sc;
    shift2[i] = beta_out[i] - m_out[i] * sc;
  }
}

// ---------------------------------------------------------------------------
// prep2: P1[o][d] = s2[o]*w_out[o][d]*s_wg[d];  c1[o]
// ---------------------------------------------------------------------------
__global__ __launch_bounds__(128) void k_prep2(
    const float* __restrict__ w_out, const float* __restrict__ b_out,
    const float* __restrict__ scale_wg, const float* __restrict__ shift_wg,
    const float* __restrict__ scale2, const float* __restrict__ shift2,
    _Float16* __restrict__ P1, float* __restrict__ c1) {
  const int o = blockIdx.x, d = threadIdx.x;
  const float wo = w_out[o * 128 + d];
  P1[o * 128 + d] = (_Float16)(scale2[o] * wo * scale_wg[d]);
  __shared__ float red[128];
  red[d] = wo * shift_wg[d];
  __syncthreads();
  for (int st = 64; st; st >>= 1) {
    if (d < st) red[d] += red[d + st];
    __syncthreads();
  }
  if (d == 0) c1[o] = scale2[o] * (red[0] + b_out[o]) + shift2[o];
}

// ---------------------------------------------------------------------------
// K1: fused QKV projection. Out tile [384 x 64] per workgroup.
//   rows 0..127   -> kT[b][n][o]  (transposed, packed 16B stores)
//   rows 128..255 -> q[b][o][n]
//   rows 256..383 -> v[b][o][n]
// ---------------------------------------------------------------------------
__global__ __launch_bounds__(256) void k_qkv(
    const float* __restrict__ x, const _Float16* __restrict__ Wall,
    const float* __restrict__ ball, _Float16* __restrict__ kT,
    _Float16* __restrict__ qb, _Float16* __restrict__ vb) {
  __shared__ _Float16 xt[64 * 264];  // x tile transposed -> [n][c], 33 KB
  const int b = blockIdx.y;
  const int n0 = blockIdx.x * 64;
  const int tid = threadIdx.x, lane = tid & 31, wave = tid >> 5;

  for (int i = tid; i < 256 * 64; i += 256) {
    int c = i >> 6, nn = i & 63;
    xt[nn * 264 + c] = (_Float16)x[((size_t)b * PLANE + c) * NPIX + n0 + nn];
  }
  __syncthreads();

#pragma unroll
  for (int rt = 0; rt < 3; ++rt) {
    const int m0 = wave * 48 + rt * 16;  // output row tile (0..383)
    v8f acc[4];
#pragma unroll
    for (int ct = 0; ct < 4; ++ct) acc[ct] = vzero8();
#pragma unroll
    for (int ks = 0; ks < 256; ks += 32) {
      v16h a = frag_a(Wall, 256, m0, ks, lane);
#pragma unroll
      for (int ct = 0; ct < 4; ++ct)
        acc[ct] = wmma_f16(a, frag_b(xt, 264, ct * 16, ks, lane), acc[ct]);
    }
    float bias[8];
#pragma unroll
    for (int r = 0; r < 8; ++r) bias[r] = ball[m0 + ((lane >> 4) << 3) + r];

    if (m0 < 128) {  // k rows -> transposed store
#pragma unroll
      for (int ct = 0; ct < 4; ++ct) {
        F16x8 pk;
#pragma unroll
        for (int r = 0; r < 8; ++r) pk.h[r] = (_Float16)(acc[ct][r] + bias[r]);
        int n = n0 + ct * 16 + (lane & 15);
        *(uint4*)(kT + ((size_t)b * NPIX + n) * 128 + m0 + ((lane >> 4) << 3)) = pk.u;
      }
    } else {  // q / v rows -> natural layout
      _Float16* dst = (m0 < 256) ? qb : vb;
      const int mb = (m0 < 256) ? (m0 - 128) : (m0 - 256);
#pragma unroll
      for (int ct = 0; ct < 4; ++ct) {
        int n = n0 + ct * 16 + (lane & 15);
#pragma unroll
        for (int r = 0; r < 8; ++r)
          dst[((size_t)b * INTER + mb + ((lane >> 4) << 3) + r) * NPIX + n] =
              (_Float16)(acc[ct][r] + bias[r]);
      }
    }
  }
}

// ---------------------------------------------------------------------------
// K2: logits partials. Spart[b][kc][c][d] = sum_{n in slice} q[c,n] v[d,n]
// grid (32, 8): each wg owns a K-slice of 512, full 128x128 output.
// q/v tiles staged via async global->LDS when available.
// ---------------------------------------------------------------------------
__global__ __launch_bounds__(256) void k_logits(
    const _Float16* __restrict__ qb, const _Float16* __restrict__ vb,
    float* __restrict__ Spart) {
  __shared__ _Float16 qs[128 * 72];  // 18 KB
  __shared__ _Float16 vs[128 * 72];  // 18 KB
  const int b = blockIdx.y, kc = blockIdx.x;
  const int tid = threadIdx.x, lane = tid & 31, wave = tid >> 5;

  v8f acc[8];
#pragma unroll
  for (int ct = 0; ct < 8; ++ct) acc[ct] = vzero8();

  for (int chunk = 0; chunk < 8; ++chunk) {
    const int k0 = kc * 512 + chunk * 64;
    __syncthreads();  // prior reads of qs/vs done before overwriting
#pragma unroll
    for (int i = tid; i < 1024; i += 256) {
      int row = i >> 3, c16 = i & 7;
      stage16((const uint4*)(qb + ((size_t)b * INTER + row) * NPIX + k0) + c16,
              qs + row * 72 + c16 * 8);
      stage16((const uint4*)(vb + ((size_t)b * INTER + row) * NPIX + k0) + c16,
              vs + row * 72 + c16 * 8);
    }
    stage_wait();
    __syncthreads();
#pragma unroll
    for (int kst = 0; kst < 64; kst += 32) {
      v16h a = frag_a(qs, 72, wave * 16, kst, lane);
#pragma unroll
      for (int ct = 0; ct < 8; ++ct)
        acc[ct] = wmma_f16(a, frag_b(vs, 72, ct * 16, kst, lane), acc[ct]);
    }
  }
  float* dst = Spart + (((size_t)b * 32 + kc) * 128 + wave * 16) * 128;
#pragma unroll
  for (int ct = 0; ct < 8; ++ct)
#pragma unroll
    for (int r = 0; r < 8; ++r)
      dst[(r + ((lane >> 4) << 3)) * 128 + ct * 16 + (lane & 15)] = acc[ct][r];
}

// ---------------------------------------------------------------------------
// K3: reduce 32 partials + row softmax over d. A[b][c][d] f16 (row-major).
// ---------------------------------------------------------------------------
__global__ __launch_bounds__(128) void k_softmax(const float* __restrict__ Spart,
                                                 _Float16* __restrict__ Ab) {
  const int b = blockIdx.y, c = blockIdx.x, d = threadIdx.x;
  float s = 0.f;
  for (int kc = 0; kc < 32; ++kc)
    s += Spart[(((size_t)b * 32 + kc) * 128 + c) * 128 + d];
  __shared__ float red[128];
  red[d] = s;
  __syncthreads();
  for (int st = 64; st; st >>= 1) {
    if (d < st) red[d] = fmaxf(red[d], red[d + st]);
    __syncthreads();
  }
  const float mx = red[0];
  __syncthreads();
  const float e = __expf(s - mx);
  red[d] = e;
  __syncthreads();
  for (int st = 64; st; st >>= 1) {
    if (d < st) red[d] += red[d + st];
    __syncthreads();
  }
  Ab[((size_t)b * 128 + c) * 128 + d] = (_Float16)(e / red[0]);
}

// ---------------------------------------------------------------------------
// K_T: T[o,c] = sum_d w_wg[o,d] * A[c,d], stored transposed Tt[c][o].
// ---------------------------------------------------------------------------
__global__ __launch_bounds__(256) void k_wgA(const _Float16* __restrict__ wwg,
                                             const _Float16* __restrict__ Ab,
                                             _Float16* __restrict__ Tt) {
  const int b = blockIdx.x;
  const int tid = threadIdx.x, lane = tid & 31, wave = tid >> 5;
  const _Float16* Arows = Ab + (size_t)b * 128 * 128;
  v8f acc[8];
#pragma unroll
  for (int ct = 0; ct < 8; ++ct) acc[ct] = vzero8();
#pragma unroll
  for (int ks = 0; ks < 128; ks += 32) {
    v16h a = frag_a(wwg, 128, wave * 16, ks, lane);
#pragma unroll
    for (int ct = 0; ct < 8; ++ct)
      acc[ct] = wmma_f16(a, frag_b(Arows, 128, ct * 16, ks, lane), acc[ct]);
  }
#pragma unroll
  for (int ct = 0; ct < 8; ++ct) {
    F16x8 pk;
#pragma unroll
    for (int r = 0; r < 8; ++r) pk.h[r] = (_Float16)acc[ct][r];
    int cc = ct * 16 + (lane & 15);
    *(uint4*)(Tt + ((size_t)b * 128 + cc) * 128 + wave * 16 + ((lane >> 4) << 3)) = pk.u;
  }
}

// ---------------------------------------------------------------------------
// K_M2: M2[p,c] = sum_o P1[p,o] * Tt[c,o]. grid (2, 8).
// ---------------------------------------------------------------------------
__global__ __launch_bounds__(256) void k_m2(const _Float16* __restrict__ P1,
                                            const _Float16* __restrict__ Tt,
                                            _Float16* __restrict__ M2) {
  const int b = blockIdx.y, ph = blockIdx.x;
  const int tid = threadIdx.x, lane = tid & 31, wave = tid >> 5;
  const int m0 = ph * 128 + wave * 16;
  const _Float16* Trows = Tt + (size_t)b * 128 * 128;
  v8f acc[8];
#pragma unroll
  for (int ct = 0; ct < 8; ++ct) acc[ct] = vzero8();
#pragma unroll
  for (int ks = 0; ks < 128; ks += 32) {
    v16h a = frag_a(P1, 128, m0, ks, lane);
#pragma unroll
    for (int ct = 0; ct < 8; ++ct)
      acc[ct] = wmma_f16(a, frag_b(Trows, 128, ct * 16, ks, lane), acc[ct]);
  }
#pragma unroll
  for (int ct = 0; ct < 8; ++ct)
#pragma unroll
    for (int r = 0; r < 8; ++r)
      M2[((size_t)b * 256 + m0 + ((lane >> 4) << 3) + r) * 128 + ct * 16 + (lane & 15)] =
          (_Float16)acc[ct][r];
}

// ---------------------------------------------------------------------------
// K4: out = relu(M2 @ k + c1 + x). grid (128, 2, 8), tile 128x128 per wg.
// kT tile staged via async global->LDS when available.
// ---------------------------------------------------------------------------
__global__ __launch_bounds__(256) void k_out(
    const _Float16* __restrict__ M2, const _Float16* __restrict__ kT,
    const float* __restrict__ c1, const float* __restrict__ x,
    float* __restrict__ out) {
  __shared__ _Float16 kts[128 * 136];  // kT tile [n][e], 34 KB
  const int b = blockIdx.z, bm = blockIdx.y;
  const int n0 = blockIdx.x * 128;
  const int tid = threadIdx.x, lane = tid & 31, wave = tid >> 5;

#pragma unroll
  for (int i = tid; i < 1024; i += 256) {
    int row = i >> 3, c16 = i & 7;
    stage16((const uint4*)(kT + ((size_t)b * NPIX + n0 + row) * 128) + c16,
            kts + row * 136 + c16 * 8);
  }
  stage_wait();
  __syncthreads();

  const int m0 = bm * 128 + wave * 16;
  const _Float16* Arows = M2 + (size_t)b * 256 * 128;
  v8f acc[8];
#pragma unroll
  for (int ct = 0; ct < 8; ++ct) acc[ct] = vzero8();
#pragma unroll
  for (int ks = 0; ks < 128; ks += 32) {
    v16h a = frag_a(Arows, 128, m0, ks, lane);
#pragma unroll
    for (int ct = 0; ct < 8; ++ct)
      acc[ct] = wmma_f16(a, frag_b(kts, 136, ct * 16, ks, lane), acc[ct]);
  }
  float c1r[8];
#pragma unroll
  for (int r = 0; r < 8; ++r) c1r[r] = c1[m0 + ((lane >> 4) << 3) + r];
#pragma unroll
  for (int ct = 0; ct < 8; ++ct) {
    const int n = n0 + ct * 16 + (lane & 15);
#pragma unroll
    for (int r = 0; r < 8; ++r) {
      const int p = m0 + ((lane >> 4) << 3) + r;
      const size_t xi = ((size_t)b * PLANE + p) * NPIX + n;
      float v = acc[ct][r] + c1r[r] + x[xi];
      out[xi] = v > 0.f ? v : 0.f;
    }
  }
}

// ---------------------------------------------------------------------------
// Workspace layout (bytes)
// ---------------------------------------------------------------------------
static constexpr size_t OFF_KT   = 0;          // f16 [8][16384][128]  33.5 MB
static constexpr size_t OFF_Q    = 33554432;   // f16 [8][128][16384]  33.5 MB
static constexpr size_t OFF_V    = 67108864;   // f16 [8][128][16384]  33.5 MB
static constexpr size_t OFF_SP   = 100663296;  // f32 [8][32][128][128] 16.8 MB
static constexpr size_t OFF_A    = 117440512;  // f16 [8][128][128]
static constexpr size_t OFF_TT   = 117702656;  // f16 [8][128][128]
static constexpr size_t OFF_M2   = 117964800;  // f16 [8][256][128]
static constexpr size_t OFF_WALL = 118489088;  // f16 [384][256]
static constexpr size_t OFF_WWG  = 118685696;  // f16 [128][128]
static constexpr size_t OFF_P1   = 118718464;  // f16 [256][128]
static constexpr size_t OFF_BALL = 118784000;  // f32 [384]
static constexpr size_t OFF_SCW  = 118785536;  // f32 [128]
static constexpr size_t OFF_SHW  = 118786048;  // f32 [128]
static constexpr size_t OFF_SC2  = 118786560;  // f32 [256]
static constexpr size_t OFF_SH2  = 118787584;  // f32 [256]
static constexpr size_t OFF_C1   = 118788608;  // f32 [256]

extern "C" void kernel_launch(void* const* d_in, const int* in_sizes, int n_in,
                              void* d_out, int out_size, void* d_ws, size_t ws_size,
                              hipStream_t stream) {
  const float* x        = (const float*)d_in[0];
  const float* wk       = (const float*)d_in[1];
  const float* bk       = (const float*)d_in[2];
  const float* wq       = (const float*)d_in[3];
  const float* bq       = (const float*)d_in[4];
  const float* wv       = (const float*)d_in[5];
  const float* bv       = (const float*)d_in[6];
  const float* w_wg     = (const float*)d_in[7];
  const float* g_wg     = (const float*)d_in[8];
  const float* beta_wg  = (const float*)d_in[9];
  const float* m_wg     = (const float*)d_in[10];
  const float* v_wg     = (const float*)d_in[11];
  const float* w_out    = (const float*)d_in[12];
  const float* b_out    = (const float*)d_in[13];
  const float* g_out    = (const float*)d_in[14];
  const float* beta_out = (const float*)d_in[15];
  const float* m_out    = (const float*)d_in[16];
  const float* v_out    = (const float*)d_in[17];
  float* out = (float*)d_out;

  char* w = (char*)d_ws;
  _Float16* kT    = (_Float16*)(w + OFF_KT);
  _Float16* qb    = (_Float16*)(w + OFF_Q);
  _Float16* vb    = (_Float16*)(w + OFF_V);
  float*    Spart = (float*)(w + OFF_SP);
  _Float16* Ab    = (_Float16*)(w + OFF_A);
  _Float16* Tt    = (_Float16*)(w + OFF_TT);
  _Float16* M2    = (_Float16*)(w + OFF_M2);
  _Float16* Wall  = (_Float16*)(w + OFF_WALL);
  _Float16* wwg16 = (_Float16*)(w + OFF_WWG);
  _Float16* P1    = (_Float16*)(w + OFF_P1);
  float* ball     = (float*)(w + OFF_BALL);
  float* scw      = (float*)(w + OFF_SCW);
  float* shw      = (float*)(w + OFF_SHW);
  float* sc2      = (float*)(w + OFF_SC2);
  float* sh2      = (float*)(w + OFF_SH2);
  float* c1       = (float*)(w + OFF_C1);

  k_prep1<<<dim3(128), 256, 0, stream>>>(wk, wq, wv, bk, bq, bv, w_wg, g_wg,
                                         beta_wg, m_wg, v_wg, g_out, beta_out,
                                         m_out, v_out, Wall, wwg16, ball, scw,
                                         shw, sc2, sh2);
  k_prep2<<<dim3(256), 128, 0, stream>>>(w_out, b_out, scw, shw, sc2, sh2, P1, c1);
  k_qkv<<<dim3(256, 8), 256, 0, stream>>>(x, Wall, ball, kT, qb, vb);
  k_logits<<<dim3(32, 8), 256, 0, stream>>>(qb, vb, Spart);
  k_softmax<<<dim3(128, 8), 128, 0, stream>>>(Spart, Ab);
  k_wgA<<<dim3(8), 256, 0, stream>>>(wwg16, Ab, Tt);
  k_m2<<<dim3(2, 8), 256, 0, stream>>>(P1, Tt, M2);
  k_out<<<dim3(128, 2, 8), 256, 0, stream>>>(M2, kT, c1, x, out);
  (void)in_sizes; (void)n_in; (void)out_size; (void)ws_size;
}
